// Agent_51324859187430
// MI455X (gfx1250) — matrix operand order repaired and checked by reading, not compiled, not run on hardware
//
#include <hip/hip_runtime.h>
#include <hip/hip_bf16.h>

typedef __attribute__((ext_vector_type(16))) _Float16 v16h;
typedef __attribute__((ext_vector_type(8)))  _Float16 v8h;
typedef _Float16 __attribute__((ext_vector_type(8), aligned(8))) v8h_a8;
typedef __attribute__((ext_vector_type(8)))  float    v8f;

// ---------------- problem dims ----------------
#define BATCH   32
#define SEQ     128
#define NBT     4096          // B*T
#define DMODEL  128
#define DSTATE  16
#define DCONV   4
#define DINNER  256
#define DTRANK  8
#define NACT    18

#define H0 44
#define C1_O 32
#define C1_K 256              // 8*8*4 (3 real ch + 1 zero pad), k = kh*32 + kw*4 + c
#define C2_O 64
#define C2_K 512              // k = (kh*4+kw)*32 + c
#define C3_O 64
#define C3_K 576              // k = (kh*3+kw)*64 + c
#define FC_K 256
#define XP_N 48               // 40 padded to 48

// ======================================================================
// WMMA fragment helpers (wave32, contiguous k order, same for A and B):
//  A lane (m = lane&15, half = lane>>4) holds K chunks
//      [k0 + 16*half, +8) and [k0 + 16*half + 8, +8)   (two 16B loads)
//  B lane (n = lane&15) holds K range [k0 + 16*half, +16) of column n
//      -> one 32B load from BT[n][K] (transposed/packed weights)
//  D VGPR j holds row m = j + 8*half, col n = lane&15
// ======================================================================
__device__ __forceinline__ v16h make_a(v8h lo, v8h hi) {
    v16h a;
#pragma unroll
    for (int e = 0; e < 8; ++e) { a[e] = lo[e]; a[8 + e] = hi[e]; }
    return a;
}

__device__ __forceinline__ v8f wmma_acc(v16h a, v16h b, v8f acc) {
    return __builtin_amdgcn_wmma_f32_16x16x32_f16(
        /*neg_a=*/false, a, /*neg_b=*/false, b,
        /*c_mod=*/(short)0, acc, /*reuse_a=*/false, /*reuse_b=*/false);
}

// generic row-major GEMM tile: A[M][K] row-major, BT[N][K] packed transposed
template <int K>
__device__ __forceinline__ v8f gemm_tile(const _Float16* __restrict__ A,
                                         const _Float16* __restrict__ BT,
                                         int m0, int n0, v8f acc) {
    const int lane = threadIdx.x & 31;
    const int mn = lane & 15, half = lane >> 4;
    const _Float16* arow = A + (size_t)(m0 + mn) * K;
    const _Float16* brow = BT + (size_t)(n0 + mn) * K;
#pragma unroll
    for (int k0 = 0; k0 < K; k0 += 32) {
        v8h lo = *(const v8h*)(arow + k0 + half * 16);
        v8h hi = *(const v8h*)(arow + k0 + 8 + half * 16);
        v16h b = *(const v16h*)(brow + k0 + half * 16);
        acc = wmma_acc(make_a(lo, hi), b, acc);
    }
    return acc;
}

__device__ __forceinline__ void store_f16_tile(_Float16* out, int ldo, const v8f& acc,
                                               const float* bias, bool relu) {
    const int lane = threadIdx.x & 31;
    const int n = lane & 15, half = lane >> 4;
    const float bb = bias ? bias[n] : 0.f;
#pragma unroll
    for (int j = 0; j < 8; ++j) {
        const int m = j + (half << 3);
        float v = acc[j] + bb;
        if (relu && v < 0.f) v = 0.f;
        out[m * ldo + n] = (_Float16)v;
    }
}

__device__ __forceinline__ void store_f32_tile(float* out, int ldo, const v8f& acc) {
    const int lane = threadIdx.x & 31;
    const int n = lane & 15, half = lane >> 4;
#pragma unroll
    for (int j = 0; j < 8; ++j) out[(j + (half << 3)) * ldo + n] = acc[j];
}

__device__ __forceinline__ float sigmoidf_(float x) { return 1.f / (1.f + __expf(-x)); }

// ======================================================================
// 0a) preprocess: x (f32 NHWC,3ch) -> xh4 (f16 NHWC, 4ch zero-padded), /255
// ======================================================================
__global__ void __launch_bounds__(256) preprocess_kernel(
    const float* __restrict__ x, _Float16* __restrict__ xh4) {
    const int i = blockIdx.x * blockDim.x + threadIdx.x;
    if (i >= NBT * H0 * H0) return;
    const float* s = x + (size_t)i * 3;
    _Float16* d = xh4 + (size_t)i * 4;
    d[0] = (_Float16)(s[0] * (1.f / 255.f));
    d[1] = (_Float16)(s[1] * (1.f / 255.f));
    d[2] = (_Float16)(s[2] * (1.f / 255.f));
    d[3] = (_Float16)0.f;
}

// ======================================================================
// 0b) weight pack: f32 -> f16, transposed BT[n][K] with im2col K orderings
// ======================================================================
__global__ void __launch_bounds__(256) pack_weights_kernel(
    const float* w1, const float* w2, const float* w3,
    const float* wfc, const float* wip, const float* wxp,
    _Float16* w1pk, _Float16* w2pk, _Float16* w3pk,
    _Float16* wfcpk, _Float16* wippk, _Float16* wxppk) {
    int i = blockIdx.x * blockDim.x + threadIdx.x;
    if (i < C1_O * C1_K) {                 // w1pk[o][k], k = kh*32 + kw*4 + c (c==3 -> 0)
        int o = i >> 8, k = i & 255;
        int c = k & 3, kw = (k >> 2) & 7, kh = k >> 5;
        w1pk[i] = (c < 3) ? (_Float16)w1[o * 192 + c * 64 + kh * 8 + kw] : (_Float16)0.f;
        return;
    }
    i -= C1_O * C1_K;
    if (i < C2_O * C2_K) {                 // w2pk[o][k], k = (kh*4+kw)*32 + c
        int o = i >> 9, k = i & 511, c = k & 31, f = k >> 5;
        w2pk[i] = (_Float16)w2[o * 512 + c * 16 + f];
        return;
    }
    i -= C2_O * C2_K;
    if (i < C3_O * C3_K) {                 // w3pk[o][k], k = (kh*3+kw)*64 + c
        int o = i / 576, k = i % 576, c = k & 63, f = k >> 6;
        w3pk[i] = (_Float16)w3[o * 576 + c * 9 + f];
        return;
    }
    i -= C3_O * C3_K;
    if (i < DMODEL * FC_K) { wfcpk[i] = (_Float16)wfc[i]; return; }      // (o,k) already
    i -= DMODEL * FC_K;
    if (i < 2 * DINNER * DMODEL) { wippk[i] = (_Float16)wip[i]; return; } // (o,k) already
    i -= 2 * DINNER * DMODEL;
    if (i < XP_N * DINNER) {               // wxppk[j][k], rows 40..47 zero
        int j = i >> 8, k = i & 255;
        wxppk[i] = (j < 40) ? (_Float16)wxp[j * 256 + k] : (_Float16)0.f;
        return;
    }
}

// ======================================================================
// 1) conv1: M=409600, K=256 (padded), N=32.  act1 NHWC f16 [n*100+p*10+q][32]
// ======================================================================
__global__ void __launch_bounds__(256) conv1_kernel(
    const _Float16* __restrict__ xh4, const _Float16* __restrict__ w1pk,
    const float* __restrict__ bias, _Float16* __restrict__ act1) {
    const int wave = (blockIdx.x * blockDim.x + threadIdx.x) >> 5;
    const int MT = (NBT * 100) / 16, NT = C1_O / 16;
    if (wave >= MT * NT) return;
    const int mt = wave / NT, nt = wave % NT;
    const int m0 = mt * 16, n0 = nt * 16;
    const int lane = threadIdx.x & 31;
    const int mn = lane & 15, half = lane >> 4;
    const int r = m0 + mn;
    const int nimg = r / 100, rem = r % 100, p = rem / 10, q = rem % 10;
    const _Float16* brow = w1pk + (size_t)(n0 + mn) * C1_K;
    v8f acc = {0.f, 0.f, 0.f, 0.f, 0.f, 0.f, 0.f, 0.f};
#pragma unroll
    for (int kh = 0; kh < 8; ++kh) {
        // pixel row (ih, iw=4q..4q+7), 4 padded f16 per pixel
        const _Float16* rowp =
            xh4 + ((size_t)(nimg * H0 + 4 * p + kh) * H0 + 4 * q) * 4;
        v8h lo = *(const v8h_a8*)(rowp + half * 16);      // kw = 4*half .. +1
        v8h hi = *(const v8h_a8*)(rowp + 8 + half * 16);  // kw = 2+4*half .. +1
        v16h b = *(const v16h*)(brow + kh * 32 + half * 16);
        acc = wmma_acc(make_a(lo, hi), b, acc);
    }
    store_f16_tile(act1 + (size_t)m0 * C1_O + n0, C1_O, acc, bias + n0, true);
}

// ======================================================================
// 2) conv2: M=65536, K=512, N=64.  act2 NHWC f16 [n*16+p*4+q][64]
// ======================================================================
__global__ void __launch_bounds__(256) conv2_kernel(
    const _Float16* __restrict__ act1, const _Float16* __restrict__ w2pk,
    const float* __restrict__ bias, _Float16* __restrict__ act2) {
    const int wave = (blockIdx.x * blockDim.x + threadIdx.x) >> 5;
    const int MT = (NBT * 16) / 16, NT = C2_O / 16;
    if (wave >= MT * NT) return;
    const int mt = wave / NT, nt = wave % NT;
    const int m0 = mt * 16, n0 = nt * 16;
    const int lane = threadIdx.x & 31;
    const int mn = lane & 15, half = lane >> 4;
    const int r = m0 + mn;
    const int nimg = r >> 4, rem = r & 15, p = rem >> 2, q = rem & 3;
    const _Float16* brow = w2pk + (size_t)(n0 + mn) * C2_K;
    __builtin_prefetch(brow, 0, 1);
    v8f acc = {0.f, 0.f, 0.f, 0.f, 0.f, 0.f, 0.f, 0.f};
#pragma unroll
    for (int f = 0; f < 16; ++f) {            // f = kh*4 + kw, one 32-ch pixel per step
        const int kh = f >> 2, kw = f & 3;
        const _Float16* pix =
            act1 + ((size_t)(nimg * 100 + (2 * p + kh) * 10 + (2 * q + kw)) << 5);
        v8h lo = *(const v8h*)(pix + half * 16);
        v8h hi = *(const v8h*)(pix + 8 + half * 16);
        v16h b = *(const v16h*)(brow + f * 32 + half * 16);
        acc = wmma_acc(make_a(lo, hi), b, acc);
    }
    store_f16_tile(act2 + (size_t)m0 * C2_O + n0, C2_O, acc, bias + n0, true);
}

// ======================================================================
// 3) conv3: M=16384, K=576, N=64.  Stores NCHW-flat: hflat[r][oc*4 + p*2+q]
// ======================================================================
__global__ void __launch_bounds__(256) conv3_kernel(
    const _Float16* __restrict__ act2, const _Float16* __restrict__ w3pk,
    const float* __restrict__ bias, _Float16* __restrict__ hflat) {
    const int wave = (blockIdx.x * blockDim.x + threadIdx.x) >> 5;
    const int MT = (NBT * 4) / 16, NT = C3_O / 16;
    if (wave >= MT * NT) return;
    const int mt = wave / NT, nt = wave % NT;
    const int m0 = mt * 16, n0 = nt * 16;
    const int lane = threadIdx.x & 31;
    const int mn = lane & 15, half = lane >> 4;
    const int r = m0 + mn;
    const int nimg = r >> 2, rem = r & 3, p = rem >> 1, q = rem & 1;
    const _Float16* brow = w3pk + (size_t)(n0 + mn) * C3_K;
    __builtin_prefetch(brow, 0, 1);
    v8f acc = {0.f, 0.f, 0.f, 0.f, 0.f, 0.f, 0.f, 0.f};
#pragma unroll
    for (int s = 0; s < 18; ++s) {            // s = f*2 + (chalf), f = kh*3+kw
        const int f = s >> 1, c0 = (s & 1) << 5;
        const int kh = f / 3, kw = f % 3;
        const _Float16* pix =
            act2 + (((size_t)(nimg * 16 + (p + kh) * 4 + (q + kw))) << 6) + c0;
        v8h lo = *(const v8h*)(pix + half * 16);
        v8h hi = *(const v8h*)(pix + 8 + half * 16);
        v16h b = *(const v16h*)(brow + s * 32 + half * 16);
        acc = wmma_acc(make_a(lo, hi), b, acc);
    }
    // scatter into NCHW flatten order expected by fc (f = c*4 + p*2 + q)
    const float bb = bias[n0 + mn];
#pragma unroll
    for (int j = 0; j < 8; ++j) {
        const int m = j + (half << 3);
        const int rr = m0 + m;
        const int ni = rr >> 2, idxp = rr & 3;
        float v = acc[j] + bb;
        if (v < 0.f) v = 0.f;
        hflat[(size_t)ni * 256 + (n0 + mn) * 4 + idxp] = (_Float16)v;
    }
}

// ======================================================================
// 4) fc: M=4096, K=256, N=128  (+bias, ReLU) -> hbuf [r][128]
// ======================================================================
__global__ void __launch_bounds__(256) fc_kernel(
    const _Float16* __restrict__ hflat, const _Float16* __restrict__ wfcpk,
    const float* __restrict__ bias, _Float16* __restrict__ hbuf) {
    const int wave = (blockIdx.x * blockDim.x + threadIdx.x) >> 5;
    const int MT = NBT / 16, NT = DMODEL / 16;
    if (wave >= MT * NT) return;
    const int mt = wave / NT, nt = wave % NT;
    const int m0 = mt * 16, n0 = nt * 16;
    v8f acc = {0.f, 0.f, 0.f, 0.f, 0.f, 0.f, 0.f, 0.f};
    acc = gemm_tile<FC_K>(hflat, wfcpk, m0, n0, acc);
    store_f16_tile(hbuf + (size_t)m0 * DMODEL + n0, DMODEL, acc, bias + n0, true);
}

// ======================================================================
// 5) in_proj: M=4096, K=128, N=512 -> xz f16 (xi | z)
// ======================================================================
__global__ void __launch_bounds__(256) inproj_kernel(
    const _Float16* __restrict__ hbuf, const _Float16* __restrict__ wippk,
    _Float16* __restrict__ xz) {
    const int wave = (blockIdx.x * blockDim.x + threadIdx.x) >> 5;
    const int MT = NBT / 16, NT = (2 * DINNER) / 16;
    if (wave >= MT * NT) return;
    const int mt = wave / NT, nt = wave % NT;
    const int m0 = mt * 16, n0 = nt * 16;
    v8f acc = {0.f, 0.f, 0.f, 0.f, 0.f, 0.f, 0.f, 0.f};
    acc = gemm_tile<DMODEL>(hbuf, wippk, m0, n0, acc);
    store_f16_tile(xz + (size_t)m0 * (2 * DINNER) + n0, 2 * DINNER, acc, nullptr, false);
}

// ======================================================================
// 6) causal depthwise conv1d (K=4) + bias + SiLU -> u (f32 + f16)
// ======================================================================
__global__ void __launch_bounds__(256) conv1d_silu_kernel(
    const _Float16* __restrict__ xz, const float* __restrict__ w,
    const float* __restrict__ bias, float* __restrict__ uf, _Float16* __restrict__ uh) {
    const int idx = blockIdx.x * blockDim.x + threadIdx.x;
    if (idx >= NBT * DINNER) return;
    const int n = idx >> 8, d = idx & 255;
    const int b = n >> 7, l = n & 127;
    float acc = bias[d];
#pragma unroll
    for (int j = 0; j < DCONV; ++j) {
        const int t = l - (DCONV - 1) + j;
        if (t >= 0) acc += w[d * DCONV + j] * (float)xz[(((size_t)(b << 7) + t) << 9) + d];
    }
    const float s = acc * sigmoidf_(acc);
    uf[idx] = s;
    uh[idx] = (_Float16)s;
}

// ======================================================================
// 7) x_proj: M=4096, K=256, N=48 (40 real) -> xdbl f32 ld=48
// ======================================================================
__global__ void __launch_bounds__(256) xproj_kernel(
    const _Float16* __restrict__ uh, const _Float16* __restrict__ wxppk,
    float* __restrict__ xdbl) {
    const int wave = (blockIdx.x * blockDim.x + threadIdx.x) >> 5;
    const int MT = NBT / 16, NT = XP_N / 16;
    if (wave >= MT * NT) return;
    const int mt = wave / NT, nt = wave % NT;
    const int m0 = mt * 16, n0 = nt * 16;
    v8f acc = {0.f, 0.f, 0.f, 0.f, 0.f, 0.f, 0.f, 0.f};
    acc = gemm_tile<DINNER>(uh, wxppk, m0, n0, acc);
    store_f32_tile(xdbl + (size_t)m0 * XP_N + n0, XP_N, acc);
}

// ======================================================================
// 8) delta = softplus(dt @ dt_proj_w.T + dt_proj_b)
// ======================================================================
__global__ void __launch_bounds__(256) delta_kernel(
    const float* __restrict__ xdbl, const float* __restrict__ dtw,
    const float* __restrict__ dtb, float* __restrict__ deltaf) {
    const int idx = blockIdx.x * blockDim.x + threadIdx.x;
    if (idx >= NBT * DINNER) return;
    const int n = idx >> 8, d = idx & 255;
    float acc = dtb[d];
#pragma unroll
    for (int r = 0; r < DTRANK; ++r) acc += xdbl[(size_t)n * XP_N + r] * dtw[d * DTRANK + r];
    deltaf[idx] = (acc > 20.f) ? acc : log1pf(__expf(acc));
}

// ======================================================================
// 9) selective scan: thread per (b,d), 16 states in regs, L=128 sequential
// ======================================================================
__global__ void __launch_bounds__(256) scan_kernel(
    const float* __restrict__ deltaf, const float* __restrict__ uf,
    const float* __restrict__ xdbl, const _Float16* __restrict__ xz,
    const float* __restrict__ A_log, const float* __restrict__ Dp,
    _Float16* __restrict__ yh) {
    const int idx = blockIdx.x * blockDim.x + threadIdx.x;
    if (idx >= BATCH * DINNER) return;
    const int b = idx >> 8, d = idx & 255;
    float Ad[DSTATE], st[DSTATE];
#pragma unroll
    for (int s = 0; s < DSTATE; ++s) {
        Ad[s] = -__expf(A_log[d * DSTATE + s]);
        st[s] = 0.f;
    }
    const float Dd = Dp[d];
    for (int t = 0; t < SEQ; ++t) {
        const int row = (b << 7) + t;
        const float de = deltaf[((size_t)row << 8) + d];
        const float uu = uf[((size_t)row << 8) + d];
        float y = 0.f;
#pragma unroll
        for (int s = 0; s < DSTATE; ++s) {
            const float Bt = xdbl[(size_t)row * XP_N + DTRANK + s];
            const float Ct = xdbl[(size_t)row * XP_N + DTRANK + DSTATE + s];
            st[s] = __expf(de * Ad[s]) * st[s] + de * Bt * uu;
            y += st[s] * Ct;
        }
        y += uu * Dd;
        const float z = (float)xz[((size_t)row << 9) + DINNER + d];
        y *= z * sigmoidf_(z);
        yh[((size_t)row << 8) + d] = (_Float16)y;
    }
}

// ======================================================================
// 10) heads on last timestep only.
//     out: [action(32) | logprob(32) | entropy(32) | value(32)] f32
// ======================================================================
__global__ void __launch_bounds__(128) head_kernel(
    const _Float16* __restrict__ yh, const float* __restrict__ opw,
    const float* __restrict__ pw, const float* __restrict__ pb,
    const float* __restrict__ aw, const float* __restrict__ ab,
    const float* __restrict__ cw, const float* __restrict__ cb,
    const int* __restrict__ action, float* __restrict__ out) {
    __shared__ float sy[DINNER];
    __shared__ float sm[DMODEL];
    __shared__ float sl[DMODEL];
    __shared__ float slog[NACT];
    __shared__ float sval;
    const int b = blockIdx.x, o = threadIdx.x;
    const int row = (b << 7) + (SEQ - 1);
    sy[o] = (float)yh[((size_t)row << 8) + o];
    sy[o + 128] = (float)yh[((size_t)row << 8) + 128 + o];
    __syncthreads();
    float acc = 0.f;
#pragma unroll 8
    for (int d = 0; d < DINNER; ++d) acc += sy[d] * opw[o * DINNER + d];
    sm[o] = acc;
    __syncthreads();
    acc = pb[o];
#pragma unroll 8
    for (int j = 0; j < DMODEL; ++j) acc += sm[j] * pw[o * DMODEL + j];
    sl[o] = acc > 0.f ? acc : 0.f;
    __syncthreads();
    if (o < NACT) {
        float lg = ab[o];
        for (int j = 0; j < DMODEL; ++j) lg += sl[j] * aw[o * DMODEL + j];
        slog[o] = lg;
    } else if (o == 32) {
        float v = cb[0];
        for (int j = 0; j < DMODEL; ++j) v += sl[j] * cw[j];
        sval = v;
    }
    __syncthreads();
    if (o == 0) {
        float mx = slog[0];
        for (int a = 1; a < NACT; ++a) mx = fmaxf(mx, slog[a]);
        float se = 0.f;
        for (int a = 0; a < NACT; ++a) se += __expf(slog[a] - mx);
        const float lse = mx + __logf(se);
        float ent = 0.f;
        for (int a = 0; a < NACT; ++a) {
            const float lp = slog[a] - lse;
            ent -= __expf(lp) * lp;
        }
        const int act = action[b];
        out[b] = (float)act;
        out[BATCH + b] = slog[act] - lse;
        out[2 * BATCH + b] = ent;
        out[3 * BATCH + b] = sval;
    }
}

// ======================================================================
// host side
// ======================================================================
extern "C" void kernel_launch(void* const* d_in, const int* in_sizes, int n_in,
                              void* d_out, int out_size, void* d_ws, size_t ws_size,
                              hipStream_t stream) {
    const float* x        = (const float*)d_in[0];
    const int*   action   = (const int*)d_in[1];
    const float* conv1_w  = (const float*)d_in[2];
    const float* conv1_b  = (const float*)d_in[3];
    const float* conv2_w  = (const float*)d_in[4];
    const float* conv2_b  = (const float*)d_in[5];
    const float* conv3_w  = (const float*)d_in[6];
    const float* conv3_b  = (const float*)d_in[7];
    const float* fc_w     = (const float*)d_in[8];
    const float* fc_b     = (const float*)d_in[9];
    const float* in_proj_w= (const float*)d_in[10];
    const float* conv1d_w = (const float*)d_in[11];
    const float* conv1d_b = (const float*)d_in[12];
    const float* x_proj_w = (const float*)d_in[13];
    const float* dt_proj_w= (const float*)d_in[14];
    const float* dt_proj_b= (const float*)d_in[15];
    const float* A_log    = (const float*)d_in[16];
    const float* Dp       = (const float*)d_in[17];
    const float* out_proj_w = (const float*)d_in[18];
    const float* post_w   = (const float*)d_in[19];
    const float* post_b   = (const float*)d_in[20];
    const float* actor_w  = (const float*)d_in[21];
    const float* actor_b  = (const float*)d_in[22];
    const float* critic_w = (const float*)d_in[23];
    const float* critic_b = (const float*)d_in[24];
    float* out = (float*)d_out;

    char* p = (char*)d_ws;
    auto alloc = [&](size_t bytes) {
        char* r = p;
        p += (bytes + 255) & ~(size_t)255;
        return r;
    };
    _Float16* w1pk  = (_Float16*)alloc((size_t)C1_O * C1_K * 2);
    _Float16* w2pk  = (_Float16*)alloc((size_t)C2_O * C2_K * 2);
    _Float16* w3pk  = (_Float16*)alloc((size_t)C3_O * C3_K * 2);
    _Float16* wfcpk = (_Float16*)alloc((size_t)DMODEL * FC_K * 2);
    _Float16* wippk = (_Float16*)alloc((size_t)2 * DINNER * DMODEL * 2);
    _Float16* wxppk = (_Float16*)alloc((size_t)XP_N * DINNER * 2);
    _Float16* xh4   = (_Float16*)alloc((size_t)NBT * H0 * H0 * 4 * 2);
    _Float16* act1  = (_Float16*)alloc((size_t)NBT * 100 * C1_O * 2);
    _Float16* act2  = (_Float16*)alloc((size_t)NBT * 16 * C2_O * 2);
    _Float16* hflat = (_Float16*)alloc((size_t)NBT * 256 * 2);
    _Float16* hbuf  = (_Float16*)alloc((size_t)NBT * DMODEL * 2);
    _Float16* xz    = (_Float16*)alloc((size_t)NBT * 2 * DINNER * 2);
    float*    uf    = (float*)alloc((size_t)NBT * DINNER * 4);
    _Float16* uh    = (_Float16*)alloc((size_t)NBT * DINNER * 2);
    float*    xdbl  = (float*)alloc((size_t)NBT * XP_N * 4);
    float*    deltaf= (float*)alloc((size_t)NBT * DINNER * 4);
    _Float16* yh    = (_Float16*)alloc((size_t)NBT * DINNER * 2);

    // 0a) preprocess x -> padded f16 NHWC
    {
        const int npix = NBT * H0 * H0;
        preprocess_kernel<<<(npix + 255) / 256, 256, 0, stream>>>(x, xh4);
    }
    // 0b) weight packing
    {
        const int total = C1_O * C1_K + C2_O * C2_K + C3_O * C3_K +
                          DMODEL * FC_K + 2 * DINNER * DMODEL + XP_N * DINNER;
        pack_weights_kernel<<<(total + 255) / 256, 256, 0, stream>>>(
            conv1_w, conv2_w, conv3_w, fc_w, in_proj_w, x_proj_w,
            w1pk, w2pk, w3pk, wfcpk, wippk, wxppk);
    }
    conv1_kernel<<<(25600 * 2) / 8, 256, 0, stream>>>(xh4, w1pk, conv1_b, act1);
    conv2_kernel<<<(4096 * 4) / 8, 256, 0, stream>>>(act1, w2pk, conv2_b, act2);
    conv3_kernel<<<(1024 * 4) / 8, 256, 0, stream>>>(act2, w3pk, conv3_b, hflat);
    fc_kernel<<<(256 * 8) / 8, 256, 0, stream>>>(hflat, wfcpk, fc_b, hbuf);
    inproj_kernel<<<(256 * 32) / 8, 256, 0, stream>>>(hbuf, wippk, xz);
    conv1d_silu_kernel<<<(NBT * DINNER) / 256, 256, 0, stream>>>(xz, conv1d_w, conv1d_b, uf, uh);
    xproj_kernel<<<(256 * 3) / 8, 256, 0, stream>>>(uh, wxppk, xdbl);
    delta_kernel<<<(NBT * DINNER) / 256, 256, 0, stream>>>(xdbl, dt_proj_w, dt_proj_b, deltaf);
    scan_kernel<<<(BATCH * DINNER) / 256, 256, 0, stream>>>(deltaf, uf, xdbl, xz, A_log, Dp, yh);
    head_kernel<<<BATCH, 128, 0, stream>>>(yh, out_proj_w, post_w, post_b,
                                           actor_w, actor_b, critic_w, critic_b,
                                           action, out);
}
